// NearestNeighborSampler_78941498900927
// MI455X (gfx1250) — matrix operand order repaired and checked by reading, not compiled, not run on hardware
//
#include <hip/hip_runtime.h>
#include <hip/hip_bf16.h>

// Problem constants (from the reference)
#define BQ    1024
#define DD    512
#define QQ    32768
#define QOLD  (QQ - BQ)            // 31744: queue[j] = queue_prev[j+BQ] for j<QOLD else x[j-QOLD]
#define NSPLIT 64                  // N-dimension splits of the column range
#define WAVES  4                   // waves per search block (4 M-tiles / block)
#define ROWS_PER_BLOCK (16 * WAVES)         // 64
#define NPART  NSPLIT              // one top-2 partial per (row, split)
#define COLS_PER_SPLIT (QQ / NSPLIT)        // 512
#define TILES_PER_SPLIT (COLS_PER_SPLIT/16) // 32

typedef __attribute__((ext_vector_type(16))) __bf16 v16bf;
typedef __attribute__((ext_vector_type(8)))  __bf16 v8bf;
typedef __attribute__((ext_vector_type(4)))  __bf16 v4bf;
typedef __attribute__((ext_vector_type(8)))  float  v8f;
typedef __attribute__((ext_vector_type(4)))  float  v4f;

// ---------------------------------------------------------------------------
// Top-2 insert with lexicographic (distance, index) order -> matches
// jax.lax.top_k tie-breaking (lower index wins on equal values). Order-
// independent, so parallel merge order does not matter.
// ---------------------------------------------------------------------------
__device__ __forceinline__ void top2_insert(float d, int j,
                                            float& d1, int& i1,
                                            float& d2, int& i2) {
  bool beats1 = (d < d1) || (d == d1 && j < i1);
  bool beats2 = (d < d2) || (d == d2 && j < i2);
  if (beats1) { d2 = d1; i2 = i1; d1 = d; i1 = j; }
  else if (beats2) { d2 = d; i2 = j; }
}

// ---------------------------------------------------------------------------
// Kernel 1: build bf16 hi/lo split of the logical queue + row norms.
// queue row j < QOLD comes from queue_prev[j+BQ], else from x[j-QOLD].
// x itself is queue[QOLD:], so the GEMM A-matrix reuses these buffers.
// ---------------------------------------------------------------------------
__global__ void __launch_bounds__(128) nns_prep(
    const float* __restrict__ x, const float* __restrict__ qprev,
    unsigned short* __restrict__ bhi_u, unsigned short* __restrict__ blo_u,
    float* __restrict__ qnorm) {
  __bf16* bhi = (__bf16*)bhi_u;
  __bf16* blo = (__bf16*)blo_u;
  const int j = blockIdx.x;
  const int t = threadIdx.x;              // 128 threads, 4 floats each
  const float* src = (j < QOLD) ? (qprev + (size_t)(j + BQ) * DD)
                                : (x + (size_t)(j - QOLD) * DD);
  const int k = t * 4;
  v4f v = *(const v4f*)(src + k);
  v4bf hi, lo;
  float s = 0.0f;
#pragma unroll
  for (int e = 0; e < 4; ++e) {
    float f  = v[e];
    __bf16 h = (__bf16)f;                 // RNE truncation
    float  r = f - (float)h;              // exact residual
    hi[e] = h;
    lo[e] = (__bf16)r;
    s += f * f;
  }
  *(v4bf*)(bhi + (size_t)j * DD + k) = hi;
  *(v4bf*)(blo + (size_t)j * DD + k) = lo;

  __shared__ float red[128];
  red[t] = s;
  __syncthreads();
  for (int off = 64; off > 0; off >>= 1) {
    if (t < off) red[t] += red[t + off];
    __syncthreads();
  }
  if (t == 0) qnorm[j] = red[0];
}

// ---------------------------------------------------------------------------
// Kernel 2: fused GEMM (bf16x3 -> f32 via v_wmma_f32_16x16x32_bf16) + top-2.
// Grid: (NSPLIT, BQ/64). Block: 128 threads = 4 waves.
// All 4 waves walk the SAME N tile t together (one barrier per tile) while
// owning 4 different M-tiles -> their B-fragment loads are identical
// addresses in the same window, so the 32KB tile panel is served 4x out of
// the WGP$ and L2-side B traffic drops ~4x. A panels are register-resident
// per wave (compiler preloads them across the fully-unrolled K loop).
//
// Fragment layouts used here (contiguous k, not the ISA 7.12.2 table order; A/B match):
//   A 16x32 bf16: lane L holds row M=L%16; K elems { (L<16?0:16)+0..15 }
//                 -> one contiguous 32B run (two 16B loads).
//   B 32x16 bf16: lane L holds col N=L%16; K elems { (L<16?0:16)+0..15 }
//                 -> one contiguous 32B run (queue rows are B columns).
//   C/D f32:      VGPR r, lanes 0-15 -> (M=r, N=lane); lanes 16-31 -> (M=8+r).
// ---------------------------------------------------------------------------
__global__ void __launch_bounds__(128) nns_search(
    const unsigned short* __restrict__ bhi_u,
    const unsigned short* __restrict__ blo_u,
    const float* __restrict__ qnorm,
    float4* __restrict__ part) {
  const __bf16* bhi = (const __bf16*)bhi_u;
  const __bf16* blo = (const __bf16*)blo_u;

  const int lane  = threadIdx.x & 31;
  const int wave  = threadIdx.x >> 5;
  const int mbase = blockIdx.y * ROWS_PER_BLOCK + wave * 16;
  const int nstart = blockIdx.x * COLS_PER_SPLIT;
  const int half  = (lane < 16) ? 0 : 1;
  const int l15   = lane & 15;

  // A-matrix = x = queue rows [QOLD, QQ)
  const __bf16* ahi = bhi + (size_t)QOLD * DD;
  const __bf16* alo = blo + (size_t)QOLD * DD;
  const int arow  = mbase + l15;          // A row held by this lane
  const int khalf = half * 16;            // A K sub-offset per half-wave
  const __bf16* aph = ahi + (size_t)arow * DD + khalf;
  const __bf16* apl = alo + (size_t)arow * DD + khalf;

  // per-lane norms of the 8 C rows this lane accumulates
  float xn[8];
#pragma unroll
  for (int r = 0; r < 8; ++r)
    xn[r] = qnorm[QOLD + mbase + half * 8 + r];

  float bd1[8], bd2[8];
  int   bi1[8], bi2[8];
#pragma unroll
  for (int r = 0; r < 8; ++r) {
    bd1[r] = 3.4e38f; bd2[r] = 3.4e38f;
    bi1[r] = 0x7fffffff; bi2[r] = 0x7fffffff;
  }

  for (int t = 0; t < TILES_PER_SPLIT; ++t) {
    // Keep the 4 waves of the block in lockstep on tile t so their identical
    // B loads coalesce in the WGP$ (4x temporal reuse of the 32KB panel).
    __syncthreads();

    const int nbase = nstart + t * 16;
    const int ncol  = nbase + l15;        // B column held by this lane
    const __bf16* bph = bhi + (size_t)ncol * DD + half * 16;
    const __bf16* bpl = blo + (size_t)ncol * DD + half * 16;

    v8f acc = {};
    for (int kc = 0; kc < DD; kc += 32) {
      v8bf a0 = *(const v8bf*)(aph + kc);
      v8bf a1 = *(const v8bf*)(aph + kc + 8);
      v16bf ah = __builtin_shufflevector(a0, a1,
          0,1,2,3,4,5,6,7,8,9,10,11,12,13,14,15);
      v8bf c0 = *(const v8bf*)(apl + kc);
      v8bf c1 = *(const v8bf*)(apl + kc + 8);
      v16bf al = __builtin_shufflevector(c0, c1,
          0,1,2,3,4,5,6,7,8,9,10,11,12,13,14,15);
      v16bf bh = *(const v16bf*)(bph + kc);
      v16bf bl = *(const v16bf*)(bpl + kc);
      // bf16x3: dot ~= lo*hi + hi*lo + hi*hi  (drop lo*lo, ~2^-16 rel err)
      acc = __builtin_amdgcn_wmma_f32_16x16x32_bf16(
          false, al, false, bh, (short)0, acc, false, false);
      acc = __builtin_amdgcn_wmma_f32_16x16x32_bf16(
          false, ah, false, bl, (short)0, acc, false, false);
      acc = __builtin_amdgcn_wmma_f32_16x16x32_bf16(
          false, ah, false, bh, (short)0, acc, false, false);
    }

    const float qn = qnorm[ncol];
#pragma unroll
    for (int r = 0; r < 8; ++r) {
      float d = fmaf(-2.0f, acc[r], xn[r] + qn);  // d2; sqrt is monotone
      top2_insert(d, ncol, bd1[r], bi1[r], bd2[r], bi2[r]);
    }
  }

  // Butterfly merge across the 16 lanes of each half (row M lives in one half)
#pragma unroll
  for (int r = 0; r < 8; ++r) {
#pragma unroll
    for (int m = 1; m <= 8; m <<= 1) {
      float od1 = __shfl_xor(bd1[r], m);
      int   oi1 = __shfl_xor(bi1[r], m);
      float od2 = __shfl_xor(bd2[r], m);
      int   oi2 = __shfl_xor(bi2[r], m);
      top2_insert(od1, oi1, bd1[r], bi1[r], bd2[r], bi2[r]);
      top2_insert(od2, oi2, bd1[r], bi1[r], bd2[r], bi2[r]);
    }
  }

  if (l15 == 0) {                          // lane 0 and lane 16 write
    const int p = blockIdx.x;              // one partial per (row, split)
#pragma unroll
    for (int r = 0; r < 8; ++r) {
      const int row = mbase + half * 8 + r;
      part[(size_t)row * NPART + p] =
          make_float4(bd1[r], __int_as_float(bi1[r]),
                      bd2[r], __int_as_float(bi2[r]));
    }
  }
}

// ---------------------------------------------------------------------------
// Kernel 3: per output row merge NPART partial top-2s, take the SECOND
// nearest index (indices[:, -1]) and gather the f32 queue row.
// ---------------------------------------------------------------------------
__global__ void __launch_bounds__(128) nns_finalize(
    const float4* __restrict__ part,
    const float* __restrict__ x, const float* __restrict__ qprev,
    float* __restrict__ out) {
  const int row = blockIdx.x;
  const int t   = threadIdx.x;            // 128 threads; first NPART load

  __shared__ float sd1[NPART], sd2[NPART];
  __shared__ int   si1[NPART], si2[NPART];
  __shared__ int   snn;

  if (t < NPART) {
    float4 p = part[(size_t)row * NPART + t];
    sd1[t] = p.x; si1[t] = __float_as_int(p.y);
    sd2[t] = p.z; si2[t] = __float_as_int(p.w);
  }
  __syncthreads();

  if (t == 0) {
    float d1 = 3.4e38f, d2 = 3.4e38f;
    int   i1 = 0x7fffffff, i2 = 0x7fffffff;
    for (int q = 0; q < NPART; ++q) {
      top2_insert(sd1[q], si1[q], d1, i1, d2, i2);
      top2_insert(sd2[q], si2[q], d1, i1, d2, i2);
    }
    snn = i2;                              // second-nearest (skips self-match)
  }
  __syncthreads();

  const int nn = snn;
  const float* src = (nn < QOLD) ? (qprev + (size_t)(nn + BQ) * DD)
                                 : (x + (size_t)(nn - QOLD) * DD);
  v4f v = *(const v4f*)(src + t * 4);
  *(v4f*)(out + (size_t)row * DD + t * 4) = v;
}

// ---------------------------------------------------------------------------
extern "C" void kernel_launch(void* const* d_in, const int* in_sizes, int n_in,
                              void* d_out, int out_size, void* d_ws, size_t ws_size,
                              hipStream_t stream) {
  const float* x     = (const float*)d_in[0];   // (1024, 512) f32
  const float* qprev = (const float*)d_in[1];   // (32768, 512) f32
  float* out = (float*)d_out;                   // (1024, 512) f32

  char* ws = (char*)d_ws;
  size_t off = 0;
  unsigned short* bhi = (unsigned short*)(ws + off); off += (size_t)QQ * DD * 2; // 32 MB
  unsigned short* blo = (unsigned short*)(ws + off); off += (size_t)QQ * DD * 2; // 32 MB
  float* qnorm        = (float*)(ws + off);          off += (size_t)QQ * 4;      // 128 KB
  float4* part        = (float4*)(ws + off);         off += (size_t)BQ * NPART * sizeof(float4); // 1 MB

  nns_prep<<<QQ, 128, 0, stream>>>(x, qprev, bhi, blo, qnorm);
  dim3 sgrid(NSPLIT, BQ / ROWS_PER_BLOCK);
  nns_search<<<sgrid, 128, 0, stream>>>(bhi, blo, qnorm, part);
  nns_finalize<<<BQ, 128, 0, stream>>>(part, x, qprev, out);
}
